// MixtralSparseMoeBlock_21251498180858
// MI455X (gfx1250) — matrix operand order repaired and checked by reading, not compiled, not run on hardware
//
#include <hip/hip_runtime.h>
#include <cstdint>
#include <cstddef>

typedef __attribute__((ext_vector_type(16))) __bf16 v16bf;
typedef __attribute__((ext_vector_type(8)))  float  v8f;

namespace {

constexpr int T = 2048;        // B*S tokens
constexpr int H = 1024;        // hidden size
constexpr int E = 8;           // experts
constexpr int F = 2816;        // ffn dim

constexpr int NKC1 = H / 32;   // 32   K-chunks (gate/up GEMM, K=H)
constexpr int NFT1 = F / 16;   // 176  N-tiles of 16 (gate/up)
constexpr int NKC2 = F / 32;   // 88   K-chunks (down GEMM, K=F)
constexpr int NFT2 = H / 16;   // 64   N-tiles of 16 (down)
constexpr int MTILE = 32;      // rows per grouped-GEMM tile (2 WMMA row tiles)
constexpr int CAP  = 4352;     // padded slot capacity: 4096 + 8*31 = 4344, rounded
constexpr int MAXTILES = CAP / MTILE;  // 136

constexpr size_t alignup(size_t x) { return (x + 255) & ~size_t(255); }

// ---- workspace layout (bytes) ----
constexpr size_t OFF_XB   = 0;
constexpr size_t SZ_XB    = (size_t)T * H * 2;
constexpr size_t OFF_W1B  = alignup(OFF_XB + SZ_XB);
constexpr size_t SZ_WB1   = (size_t)E * NKC1 * NFT1 * 512 * 2;   // bf16 fragment blocks
constexpr size_t OFF_W3B  = alignup(OFF_W1B + SZ_WB1);
constexpr size_t OFF_W2B  = alignup(OFF_W3B + SZ_WB1);
constexpr size_t SZ_WB2   = (size_t)E * NKC2 * NFT2 * 512 * 2;
constexpr size_t OFF_TOPI = alignup(OFF_W2B + SZ_WB2);
constexpr size_t OFF_TOPW = alignup(OFF_TOPI + (size_t)T * 2 * 4);
constexpr size_t OFF_TOK  = alignup(OFF_TOPW + (size_t)T * 2 * 4);
constexpr size_t OFF_WSL  = alignup(OFF_TOK  + (size_t)CAP * 4);
constexpr size_t OFF_SLOT = alignup(OFF_WSL  + (size_t)CAP * 4);
constexpr size_t OFF_TLE  = alignup(OFF_SLOT + (size_t)T * 2 * 4);
constexpr size_t OFF_TLB  = alignup(OFF_TLE  + (size_t)MAXTILES * 4);
constexpr size_t OFF_NT   = alignup(OFF_TLB  + (size_t)MAXTILES * 4);
constexpr size_t OFF_ACT  = alignup(OFF_NT   + 256);
constexpr size_t OFF_CON  = alignup(OFF_ACT  + (size_t)CAP * F * 2);
// total = OFF_CON + CAP*H*4  (~185 MB)

__device__ __forceinline__ uint16_t f2bf(float f) {
  union { float f; uint32_t u; } c; c.f = f;
  const uint32_t u = c.u;
  return (uint16_t)((u + 0x7FFFu + ((u >> 16) & 1u)) >> 16);  // RNE
}

union Afrag { uint4 u[2]; v16bf v; };
union Cfrag { v8f v; float f[8]; };

// ---------------- K0: x -> bf16 row-major ----------------
__global__ void k_convert_x(const float* __restrict__ x, uint32_t* __restrict__ xb2) {
  const int i = blockIdx.x * 256 + threadIdx.x;       // over T*H/4 (exact grid)
  const float4 v = ((const float4*)x)[i];
  xb2[2 * i]     = (uint32_t)f2bf(v.x) | ((uint32_t)f2bf(v.y) << 16);
  xb2[2 * i + 1] = (uint32_t)f2bf(v.z) | ((uint32_t)f2bf(v.w) << 16);
}

// ---------------- K0b: weight fp32 [E][K][N] -> bf16 WMMA B-fragment blocks ----
// block id = (e*nFt + ft)*nKc + kc ; each block is 512 bf16 = per-lane 16 elems.
// lane holds column n = ft*16 + (lane&15); element j -> k = kc*32 + (lane>>4)*8 + j (j<8), kc*32 + 16 + (lane>>4)*8 + (j-8) (j>=8).
__global__ void k_swizzle(const float* __restrict__ in, uint16_t* __restrict__ out,
                          int K, int N, int nKc, int nFt) {
  const size_t i = (size_t)blockIdx.x * 256 + threadIdx.x;  // over E*K*N (exact grid)
  const int j    = (int)(i & 15);
  const int lane = (int)((i >> 4) & 31);
  const size_t bid = i >> 9;
  const int kc = (int)(bid % (size_t)nKc);
  const size_t rest = bid / (size_t)nKc;
  const int ft = (int)(rest % (size_t)nFt);
  const int e  = (int)(rest / (size_t)nFt);
  const int n = ft * 16 + (lane & 15);
  const int k = kc * 32 + (lane >> 4) * 8 + (j < 8 ? j : 16 + (j - 8));
  out[i] = f2bf(in[((size_t)e * K + k) * N + n]);
}

// ---------------- K1: router (logits, softmax, top-2, renorm) ----------------
__global__ void __launch_bounds__(256) k_router(
    const float* __restrict__ x, const float* __restrict__ gw,
    float* __restrict__ logits, int* __restrict__ topi, float* __restrict__ topw) {
  __shared__ float4 gws[E * (H / 4)];
  for (int i = threadIdx.x; i < E * (H / 4); i += 256)
    gws[i] = ((const float4*)gw)[i];
  __syncthreads();
  const int t = blockIdx.x * 256 + threadIdx.x;
  const float4* xr = (const float4*)(x + (size_t)t * H);
  float acc[E] = {};
  for (int h4 = 0; h4 < H / 4; ++h4) {
    const float4 xv = xr[h4];
    #pragma unroll
    for (int e = 0; e < E; ++e) {
      const float4 g = gws[e * (H / 4) + h4];
      acc[e] += xv.x * g.x + xv.y * g.y + xv.z * g.z + xv.w * g.w;
    }
  }
  float mx = acc[0];
  #pragma unroll
  for (int e = 1; e < E; ++e) mx = fmaxf(mx, acc[e]);
  float p[E]; float s = 0.f;
  #pragma unroll
  for (int e = 0; e < E; ++e) { p[e] = __expf(acc[e] - mx); s += p[e]; }
  const float inv = 1.0f / s;
  #pragma unroll
  for (int e = 0; e < E; ++e) { logits[(size_t)t * E + e] = acc[e]; p[e] *= inv; }
  int i0 = 0; float p0 = p[0];
  #pragma unroll
  for (int e = 1; e < E; ++e) if (p[e] > p0) { p0 = p[e]; i0 = e; }
  int i1 = (i0 == 0) ? 1 : 0; float p1 = p[i1];
  #pragma unroll
  for (int e = 0; e < E; ++e) if (e != i0 && p[e] > p1) { p1 = p[e]; i1 = e; }
  const float r = 1.0f / (p0 + p1);
  topi[2 * t] = i0; topi[2 * t + 1] = i1;
  topw[2 * t] = p0 * r; topw[2 * t + 1] = p1 * r;
}

// ---------------- K2: build per-expert token lists (deterministic) ----------
__global__ void k_route_build(const int* __restrict__ topi, const float* __restrict__ topw,
                              int* __restrict__ tok, float* __restrict__ wsl,
                              int* __restrict__ slot_of, int* __restrict__ tile_e,
                              int* __restrict__ tile_b, int* __restrict__ ntiles) {
  __shared__ int cnt[E];
  __shared__ int off[E];
  const int tid = threadIdx.x;
  if (tid < E) {
    int c = 0;
    for (int t = 0; t < T; ++t) {
      if (topi[2 * t] == tid) ++c;
      if (topi[2 * t + 1] == tid) ++c;
    }
    cnt[tid] = c;
  }
  __syncthreads();
  if (tid == 0) {
    int o = 0, tiles = 0;
    for (int e = 0; e < E; ++e) {
      off[e] = o;
      const int pad = (cnt[e] + (MTILE - 1)) & ~(MTILE - 1);
      for (int b = 0; b < pad; b += MTILE) { tile_e[tiles] = e; tile_b[tiles] = o + b; ++tiles; }
      o += pad;
    }
    *ntiles = tiles;
  }
  __syncthreads();
  if (tid < E) {
    const int o = off[tid];
    int j = 0;
    for (int t = 0; t < T; ++t) {
      #pragma unroll
      for (int k = 0; k < 2; ++k) {
        if (topi[2 * t + k] == tid) {
          tok[o + j] = t;
          wsl[o + j] = topw[2 * t + k];
          slot_of[2 * t + k] = o + j;
          ++j;
        }
      }
    }
    const int pad = (cnt[tid] + (MTILE - 1)) & ~(MTILE - 1);
    for (; j < pad; ++j) { tok[o + j] = 0; wsl[o + j] = 0.0f; }  // harmless pad rows
  }
}

// ---------------- K3: grouped gate/up GEMM + SiLU (bf16 WMMA, M=32) --------
// grid: (MAXTILES, F/128), block 256 (8 waves). Wave w -> 16 F-cols, 32 rows.
// Each B fragment is reused by two WMMAs (two 16-row A tiles).
__global__ void __launch_bounds__(256) k_gateup(
    const uint16_t* __restrict__ xb, const uint16_t* __restrict__ w1b,
    const uint16_t* __restrict__ w3b, const int* __restrict__ tok,
    const float* __restrict__ wsl, const int* __restrict__ tile_e,
    const int* __restrict__ tile_b, const int* __restrict__ ntiles,
    uint16_t* __restrict__ act) {
  if ((int)blockIdx.x >= *ntiles) return;
  const int e  = tile_e[blockIdx.x];
  const int tb = tile_b[blockIdx.x];
  const int wave = threadIdx.x >> 5;
  const int lane = threadIdx.x & 31;
  const int m  = lane & 15;       // A row / C column within row tile
  const int kh = lane >> 4;       // K-half select
  const int f0 = blockIdx.y * 128 + wave * 16;
  const int ft = f0 >> 4;

  const int t0 = tok[tb + m];
  const int t1 = tok[tb + 16 + m];
  const uint16_t* arow0 = xb + (size_t)t0 * H + kh * 8;
  const uint16_t* arow1 = xb + (size_t)t1 * H + kh * 8;
  const uint16_t* b1 = w1b + (((size_t)e * NFT1 + ft) * NKC1) * 512 + lane * 16;
  const uint16_t* b3 = w3b + (((size_t)e * NFT1 + ft) * NKC1) * 512 + lane * 16;

  v8f g0 = {}, g1 = {}, u0 = {}, u1 = {};
  #pragma unroll 4
  for (int kc = 0; kc < NKC1; ++kc) {
    Afrag A0, A1, B1, B3;
    A0.u[0] = *(const uint4*)(arow0);
    A0.u[1] = *(const uint4*)(arow0 + 16);
    A1.u[0] = *(const uint4*)(arow1);
    A1.u[1] = *(const uint4*)(arow1 + 16);
    arow0 += 32; arow1 += 32;
    B1.u[0] = *(const uint4*)(b1);
    B1.u[1] = *(const uint4*)(b1 + 8);
    B3.u[0] = *(const uint4*)(b3);
    B3.u[1] = *(const uint4*)(b3 + 8);
    b1 += 512; b3 += 512;
    g0 = __builtin_amdgcn_wmma_f32_16x16x32_bf16(false, A0.v, false, B1.v, (short)0, g0, false, false);
    g1 = __builtin_amdgcn_wmma_f32_16x16x32_bf16(false, A1.v, false, B1.v, (short)0, g1, false, false);
    u0 = __builtin_amdgcn_wmma_f32_16x16x32_bf16(false, A0.v, false, B3.v, (short)0, u0, false, false);
    u1 = __builtin_amdgcn_wmma_f32_16x16x32_bf16(false, A1.v, false, B3.v, (short)0, u1, false, false);
  }
  {
    Cfrag G, U; G.v = g0; U.v = u0;
    uint16_t* dst = act + (size_t)(tb + kh * 8) * F + f0 + m;
    #pragma unroll
    for (int i = 0; i < 8; ++i) {
      const float w  = wsl[tb + kh * 8 + i];
      const float gv = G.f[i];
      const float sv = gv / (1.0f + __expf(-gv));   // SiLU
      dst[(size_t)i * F] = f2bf(sv * U.f[i] * w);
    }
  }
  {
    Cfrag G, U; G.v = g1; U.v = u1;
    uint16_t* dst = act + (size_t)(tb + 16 + kh * 8) * F + f0 + m;
    #pragma unroll
    for (int i = 0; i < 8; ++i) {
      const float w  = wsl[tb + 16 + kh * 8 + i];
      const float gv = G.f[i];
      const float sv = gv / (1.0f + __expf(-gv));   // SiLU
      dst[(size_t)i * F] = f2bf(sv * U.f[i] * w);
    }
  }
}

// ---------------- K4: grouped down GEMM (bf16 WMMA, M=32) ----------------
// grid: (MAXTILES, H/128), block 256 (8 waves).
__global__ void __launch_bounds__(256) k_down(
    const uint16_t* __restrict__ act, const uint16_t* __restrict__ w2b,
    const int* __restrict__ tile_e, const int* __restrict__ tile_b,
    const int* __restrict__ ntiles, float* __restrict__ con) {
  if ((int)blockIdx.x >= *ntiles) return;
  const int e  = tile_e[blockIdx.x];
  const int tb = tile_b[blockIdx.x];
  const int wave = threadIdx.x >> 5;
  const int lane = threadIdx.x & 31;
  const int m  = lane & 15;
  const int kh = lane >> 4;
  const int h0 = blockIdx.y * 128 + wave * 16;
  const int ht = h0 >> 4;

  const uint16_t* arow0 = act + (size_t)(tb + m) * F + kh * 8;
  const uint16_t* arow1 = act + (size_t)(tb + 16 + m) * F + kh * 8;
  const uint16_t* b2 = w2b + (((size_t)e * NFT2 + ht) * NKC2) * 512 + lane * 16;

  v8f c0 = {}, c1 = {};
  #pragma unroll 4
  for (int kc = 0; kc < NKC2; ++kc) {
    Afrag A0, A1, Bf;
    A0.u[0] = *(const uint4*)(arow0);
    A0.u[1] = *(const uint4*)(arow0 + 16);
    A1.u[0] = *(const uint4*)(arow1);
    A1.u[1] = *(const uint4*)(arow1 + 16);
    arow0 += 32; arow1 += 32;
    Bf.u[0] = *(const uint4*)(b2);
    Bf.u[1] = *(const uint4*)(b2 + 8);
    b2 += 512;
    c0 = __builtin_amdgcn_wmma_f32_16x16x32_bf16(false, A0.v, false, Bf.v, (short)0, c0, false, false);
    c1 = __builtin_amdgcn_wmma_f32_16x16x32_bf16(false, A1.v, false, Bf.v, (short)0, c1, false, false);
  }
  {
    Cfrag C; C.v = c0;
    float* dst = con + (size_t)(tb + kh * 8) * H + h0 + m;
    #pragma unroll
    for (int i = 0; i < 8; ++i) dst[(size_t)i * H] = C.f[i];
  }
  {
    Cfrag C; C.v = c1;
    float* dst = con + (size_t)(tb + 16 + kh * 8) * H + h0 + m;
    #pragma unroll
    for (int i = 0; i < 8; ++i) dst[(size_t)i * H] = C.f[i];
  }
}

// ---------------- K5: final combine (two slots per token) ----------------
__global__ void k_combine(const float* __restrict__ con, const int* __restrict__ slot_of,
                          float* __restrict__ out) {
  const int i = blockIdx.x * 256 + threadIdx.x;       // over T*H/4 (exact grid)
  const int t  = i / (H / 4);
  const int h4 = i % (H / 4);
  const int s0 = slot_of[2 * t];
  const int s1 = slot_of[2 * t + 1];
  const float4 a = ((const float4*)con)[(size_t)s0 * (H / 4) + h4];
  const float4 b = ((const float4*)con)[(size_t)s1 * (H / 4) + h4];
  float4 r; r.x = a.x + b.x; r.y = a.y + b.y; r.z = a.z + b.z; r.w = a.w + b.w;
  ((float4*)out)[i] = r;
}

}  // namespace

extern "C" void kernel_launch(void* const* d_in, const int* in_sizes, int n_in,
                              void* d_out, int out_size, void* d_ws, size_t ws_size,
                              hipStream_t stream) {
  (void)in_sizes; (void)n_in; (void)out_size; (void)ws_size;
  const float* x  = (const float*)d_in[0];   // [T, H]
  const float* gw = (const float*)d_in[1];   // [E, H]
  const float* w1 = (const float*)d_in[2];   // [E, H, F]
  const float* w3 = (const float*)d_in[3];   // [E, H, F]
  const float* w2 = (const float*)d_in[4];   // [E, F, H]
  float* out = (float*)d_out;                // [T*H] final | [T*E] logits

  char* ws = (char*)d_ws;
  uint16_t* xb   = (uint16_t*)(ws + OFF_XB);
  uint16_t* w1b  = (uint16_t*)(ws + OFF_W1B);
  uint16_t* w3b  = (uint16_t*)(ws + OFF_W3B);
  uint16_t* w2b  = (uint16_t*)(ws + OFF_W2B);
  int*      topi = (int*)(ws + OFF_TOPI);
  float*    topw = (float*)(ws + OFF_TOPW);
  int*      tok  = (int*)(ws + OFF_TOK);
  float*    wsl  = (float*)(ws + OFF_WSL);
  int*      slot = (int*)(ws + OFF_SLOT);
  int*      tle  = (int*)(ws + OFF_TLE);
  int*      tlb  = (int*)(ws + OFF_TLB);
  int*      nt   = (int*)(ws + OFF_NT);
  uint16_t* act  = (uint16_t*)(ws + OFF_ACT);
  float*    con  = (float*)(ws + OFF_CON);

  const int wtot_blocks = (int)(((size_t)E * H * F) / 256);   // exact

  k_convert_x<<<(T * H / 4) / 256, 256, 0, stream>>>(x, (uint32_t*)xb);
  k_swizzle<<<wtot_blocks, 256, 0, stream>>>(w1, w1b, H, F, NKC1, NFT1);
  k_swizzle<<<wtot_blocks, 256, 0, stream>>>(w3, w3b, H, F, NKC1, NFT1);
  k_swizzle<<<wtot_blocks, 256, 0, stream>>>(w2, w2b, F, H, NKC2, NFT2);
  k_router<<<T / 256, 256, 0, stream>>>(x, gw, out + (size_t)T * H, topi, topw);
  k_route_build<<<1, 32, 0, stream>>>(topi, topw, tok, wsl, slot, tle, tlb, nt);
  k_gateup<<<dim3(MAXTILES, F / 128), 256, 0, stream>>>(xb, w1b, w3b, tok, wsl, tle, tlb, nt, act);
  k_down<<<dim3(MAXTILES, H / 128), 256, 0, stream>>>(act, w2b, tle, tlb, nt, con);
  k_combine<<<(T * H / 4) / 256, 256, 0, stream>>>(con, slot, out);
}
